// MaskedCrossAttention_24610162606215
// MI455X (gfx1250) — matrix operand order repaired and checked by reading, not compiled, not run on hardware
//
#include <hip/hip_runtime.h>

// ---------------------------------------------------------------------------
// MaskedCrossAttention for MI455X (gfx1250, wave32, WMMA)
//
// Dominant cost = 3 GEMMs (~40 GFLOP) -> bf16 WMMA (v_wmma_f32_16x16x32_bf16)
// with f32 accumulation. Attention itself is 64-keys-per-query (mask picks one
// image) -> cheap VALU wave kernel.
// ---------------------------------------------------------------------------

typedef __attribute__((ext_vector_type(16))) __bf16 v16bf;
typedef __attribute__((ext_vector_type(8)))  float  v8f;

#define B_       4
#define T_TXT_   2048
#define T_IMG_   8
#define N_LAT_   64
#define DIM_     2048
#define DIM_VIS_ 1024
#define HEADS_   8
#define DHEAD_   64
#define INNER_   512   // HEADS_*DHEAD_
#define ROWS_    (B_ * T_TXT_)       // 8192
#define JKEYS_   (T_IMG_ * N_LAT_)   // 512 media tokens per batch

__device__ __forceinline__ __bf16 f2bf(float f) {
  // round-to-nearest-even f32 -> bf16
  unsigned int u = __float_as_uint(f);
  unsigned int r = (u + 0x7FFFu + ((u >> 16) & 1u)) >> 16;
  unsigned short s = (unsigned short)r;
  return __builtin_bit_cast(__bf16, s);
}

// ---------------------------------------------------------------------------
// LayerNorm (row of 2048) fused with bf16 conversion
// ---------------------------------------------------------------------------
__global__ void __launch_bounds__(256) ln_bf16_kernel(
    const float* __restrict__ x, const float* __restrict__ gamma,
    const float* __restrict__ beta, __bf16* __restrict__ xn) {
  const int row = blockIdx.x;
  const int tid = threadIdx.x;
  const float* xr = x + (size_t)row * DIM_;
  float v[8];
  float s = 0.f;
#pragma unroll
  for (int e = 0; e < 8; ++e) { v[e] = xr[e * 256 + tid]; s += v[e]; }
  __shared__ float red[256];
  red[tid] = s;
  __syncthreads();
  for (int off = 128; off > 0; off >>= 1) {
    if (tid < off) red[tid] += red[tid + off];
    __syncthreads();
  }
  const float mu = red[0] * (1.0f / (float)DIM_);
  __syncthreads();
  float vs = 0.f;
#pragma unroll
  for (int e = 0; e < 8; ++e) { float d = v[e] - mu; vs += d * d; }
  red[tid] = vs;
  __syncthreads();
  for (int off = 128; off > 0; off >>= 1) {
    if (tid < off) red[tid] += red[tid + off];
    __syncthreads();
  }
  const float rs = rsqrtf(red[0] * (1.0f / (float)DIM_) + 1e-5f);
  __bf16* xo = xn + (size_t)row * DIM_;
#pragma unroll
  for (int e = 0; e < 8; ++e) {
    const int c = e * 256 + tid;
    xo[c] = f2bf((v[e] - mu) * rs * gamma[c] + beta[c]);
  }
}

// ---------------------------------------------------------------------------
// Elementwise f32 -> bf16
// ---------------------------------------------------------------------------
__global__ void __launch_bounds__(256) conv_bf16_kernel(
    const float* __restrict__ in, __bf16* __restrict__ out, int n) {
  const int idx = blockIdx.x * 256 + threadIdx.x;
  if (idx < n) out[idx] = f2bf(in[idx]);
}

// ---------------------------------------------------------------------------
// Transposing f32 (R x C) -> bf16 (C x R); weights become N x K row-major so
// B-matrix WMMA fragments are contiguous dword loads.
// ---------------------------------------------------------------------------
__global__ void __launch_bounds__(256) conv_transpose_kernel(
    const float* __restrict__ in, __bf16* __restrict__ out, int R, int C) {
  const int idx = blockIdx.x * 256 + threadIdx.x;
  const int r = idx / C;
  const int c = idx - r * C;
  out[(size_t)c * R + r] = f2bf(in[idx]);
}

// ---------------------------------------------------------------------------
// text_time = inclusive cumsum of media_locations (per batch row of 2048)
// 1024 threads, 2 elements each, Hillis-Steele scan in LDS.
// ---------------------------------------------------------------------------
__global__ void __launch_bounds__(1024) scan_kernel(
    const unsigned char* __restrict__ loc, int* __restrict__ tt) {
  const int b = blockIdx.x;
  const int t = threadIdx.x;
  const unsigned char* L = loc + (size_t)b * T_TXT_;
  const int a0 = L[2 * t] ? 1 : 0;
  const int a1 = L[2 * t + 1] ? 1 : 0;
  const int val = a0 + a1;
  __shared__ int sh[1024];
  sh[t] = val;
  __syncthreads();
  for (int off = 1; off < 1024; off <<= 1) {
    const int add = (t >= off) ? sh[t - off] : 0;
    __syncthreads();
    sh[t] += add;
    __syncthreads();
  }
  const int excl = sh[t] - val;
  tt[(size_t)b * T_TXT_ + 2 * t]     = excl + a0;
  tt[(size_t)b * T_TXT_ + 2 * t + 1] = excl + a0 + a1;
}

// ---------------------------------------------------------------------------
// WMMA fragment loaders (this code's contiguous k order, same for A and B)
// A (16x32 bf16, row-major MxK):
//   lane = half*16 + m; VGPR r: K = half*16 + 2r (+1)
//   -> dword offsets within the row: half*8 + r
// B (32x16 bf16) loaded from TRANSPOSED storage (N x K row-major):
//   lane = half*16 + n; VGPR r: K = half*16 + 2r (+1)  -> dword half*8 + r
// ---------------------------------------------------------------------------
__device__ __forceinline__ v16bf load_fragA(const __bf16* __restrict__ A,
                                            int K, int row, int k0,
                                            int half, int ml) {
  const unsigned int* p =
      reinterpret_cast<const unsigned int*>(A + (size_t)(row + ml) * K + k0);
  union { v16bf v; unsigned int u[8]; } f;
#pragma unroll
  for (int r = 0; r < 8; ++r) {
    const int dw = half * 8 + r;
    f.u[r] = p[dw];
  }
  return f.v;
}

__device__ __forceinline__ v16bf load_fragB(const __bf16* __restrict__ BT,
                                            int K, int col, int k0,
                                            int half, int nl) {
  const unsigned int* p =
      reinterpret_cast<const unsigned int*>(BT + (size_t)(col + nl) * K + k0);
  union { v16bf v; unsigned int u[8]; } f;
#pragma unroll
  for (int r = 0; r < 8; ++r) f.u[r] = p[half * 8 + r];
  return f.v;
}

// ---------------------------------------------------------------------------
// bf16 WMMA GEMM: C(MxN,f32) = A(MxK,bf16 row-major) * BT(NxK,bf16 row-major)^T
// Block = 128 threads = 4 waves; each wave owns a 64x64 tile (4x4 WMMA frags).
// Grid: (N/128, M/128). Requires M%128==0, N%128==0, K%32==0.
// ---------------------------------------------------------------------------
__global__ void __launch_bounds__(128) gemm_bf16_wmma(
    const __bf16* __restrict__ A, const __bf16* __restrict__ BT,
    float* __restrict__ C, int M, int N, int K) {
  const int lane = threadIdx.x & 31;
  const int wave = threadIdx.x >> 5;
  const int m0 = blockIdx.y * 128 + (wave >> 1) * 64;
  const int n0 = blockIdx.x * 128 + (wave & 1) * 64;
  const int half = (lane >> 4) & 1;
  const int ml = lane & 15;

  v8f acc[4][4] = {};  // zero-initialized accumulators

  for (int k0 = 0; k0 < K; k0 += 32) {
    v16bf af[4], bfr[4];
#pragma unroll
    for (int i = 0; i < 4; ++i) af[i]  = load_fragA(A,  K, m0 + 16 * i, k0, half, ml);
#pragma unroll
    for (int j = 0; j < 4; ++j) bfr[j] = load_fragB(BT, K, n0 + 16 * j, k0, half, ml);
#pragma unroll
    for (int i = 0; i < 4; ++i)
#pragma unroll
      for (int j = 0; j < 4; ++j)
        acc[i][j] = __builtin_amdgcn_wmma_f32_16x16x32_bf16(
            false, af[i], false, bfr[j], (short)0, acc[i][j], false, false);
  }

  // C/D layout: VGPR r -> row = 16*i + half*8 + r ; col = 16*j + (lane&15)
#pragma unroll
  for (int i = 0; i < 4; ++i)
#pragma unroll
    for (int j = 0; j < 4; ++j)
#pragma unroll
      for (int r = 0; r < 8; ++r) {
        const int rr = m0 + 16 * i + half * 8 + r;
        const int cc = n0 + 16 * j + ml;
        C[(size_t)rr * N + cc] = acc[i][j][r];
      }
}

// ---------------------------------------------------------------------------
// Attention core. Mask structure: query i attends only to the 64 latents of
// image (text_time[i]-1); zero output if text_time==0. One wave per (b,i,h),
// lane handles 2 keys / 2 output dims. Uniform control flow across the block
// so the __syncthreads are safe.
// ---------------------------------------------------------------------------
__global__ void __launch_bounds__(256) attn_core_kernel(
    const float* __restrict__ q, const float* __restrict__ kv,
    const int* __restrict__ tt, __bf16* __restrict__ ao) {
  const int lane = threadIdx.x & 31;
  const int w = threadIdx.x >> 5;
  const int gw = blockIdx.x * 8 + w;       // (row*HEADS + h)
  const int h = gw & (HEADS_ - 1);
  const int row = gw >> 3;                 // b*T_TXT + i
  const int b = row >> 11;                 // / T_TXT_

  __shared__ float sq[8][64];
  __shared__ float sp[8][64];

  const float* qr = q + (size_t)row * INNER_ + h * DHEAD_;
  sq[w][lane]      = qr[lane] * 0.125f;        // DIM_HEAD^-0.5
  sq[w][lane + 32] = qr[lane + 32] * 0.125f;
  __syncthreads();

  const int t = tt[row];
  const int ts = (t > 0 ? t : 1) - 1;          // clamped image index
  const float* kb = kv + ((size_t)b * JKEYS_ + (size_t)ts * N_LAT_) * (2 * INNER_) + h * DHEAD_;
  const float* vb = kb + INNER_;               // v cols sit 512 after k cols

  // scores for keys j=lane, lane+32
  float s0 = 0.f, s1 = 0.f;
  const float* k0r = kb + (size_t)lane * (2 * INNER_);
  const float* k1r = kb + (size_t)(lane + 32) * (2 * INNER_);
#pragma unroll 8
  for (int d = 0; d < 64; ++d) {
    const float qd = sq[w][d];
    s0 = fmaf(qd, k0r[d], s0);
    s1 = fmaf(qd, k1r[d], s1);
  }

  float m = fmaxf(s0, s1);
#pragma unroll
  for (int off = 16; off > 0; off >>= 1) m = fmaxf(m, __shfl_xor(m, off, 32));
  const float e0 = __expf(s0 - m);
  const float e1 = __expf(s1 - m);
  float sum = e0 + e1;
#pragma unroll
  for (int off = 16; off > 0; off >>= 1) sum += __shfl_xor(sum, off, 32);
  const float inv = 1.0f / sum;
  sp[w][lane]      = e0 * inv;
  sp[w][lane + 32] = e1 * inv;
  __syncthreads();

  // out[d] = sum_j p_j * v[j][d] for d = lane, lane+32
  float a0 = 0.f, a1 = 0.f;
  for (int j = 0; j < 64; ++j) {
    const float p = sp[w][j];
    const float* vr = vb + (size_t)j * (2 * INNER_);
    a0 = fmaf(p, vr[lane], a0);
    a1 = fmaf(p, vr[lane + 32], a1);
  }

  __bf16* o = ao + (size_t)row * INNER_ + h * DHEAD_;
  o[lane]      = f2bf(t ? a0 : 0.0f);
  o[lane + 32] = f2bf(t ? a1 : 0.0f);
}

// ---------------------------------------------------------------------------
// Launcher
// ---------------------------------------------------------------------------
extern "C" void kernel_launch(void* const* d_in, const int* in_sizes, int n_in,
                              void* d_out, int out_size, void* d_ws, size_t ws_size,
                              hipStream_t stream) {
  (void)in_sizes; (void)n_in; (void)out_size; (void)ws_size;
  const float* x            = (const float*)d_in[0];
  const float* media        = (const float*)d_in[1];
  const unsigned char* mloc = (const unsigned char*)d_in[2];  // jnp bool -> 1 byte
  const float* gamma        = (const float*)d_in[3];
  const float* beta         = (const float*)d_in[4];
  const float* Wq           = (const float*)d_in[5];
  const float* Wkv          = (const float*)d_in[6];
  const float* Wout         = (const float*)d_in[7];
  float* out = (float*)d_out;

  char* ws = (char*)d_ws;
  size_t off = 0;
  auto alloc = [&](size_t bytes) -> void* {
    void* p = ws + off;
    off += (bytes + 255) & ~(size_t)255;
    return p;
  };
  __bf16* xn     = (__bf16*)alloc((size_t)ROWS_ * DIM_ * 2);        // 32 MB
  __bf16* mediab = (__bf16*)alloc((size_t)B_ * JKEYS_ * DIM_VIS_ * 2); // 4 MB
  __bf16* WqT    = (__bf16*)alloc((size_t)INNER_ * DIM_ * 2);       // 2 MB
  __bf16* WkvT   = (__bf16*)alloc((size_t)(2 * INNER_) * DIM_VIS_ * 2); // 2 MB
  __bf16* WoutT  = (__bf16*)alloc((size_t)DIM_ * INNER_ * 2);       // 2 MB
  float*  qbuf   = (float*)alloc((size_t)ROWS_ * INNER_ * 4);       // 16 MB
  float*  kvbuf  = (float*)alloc((size_t)B_ * JKEYS_ * (2 * INNER_) * 4); // 8 MB
  __bf16* ao     = (__bf16*)alloc((size_t)ROWS_ * INNER_ * 2);      // 8 MB
  int*    tt     = (int*)alloc((size_t)ROWS_ * 4);

  // Pre-processing
  ln_bf16_kernel<<<ROWS_, 256, 0, stream>>>(x, gamma, beta, xn);
  conv_bf16_kernel<<<(B_ * JKEYS_ * DIM_VIS_) / 256, 256, 0, stream>>>(
      media, mediab, B_ * JKEYS_ * DIM_VIS_);
  conv_transpose_kernel<<<(DIM_ * INNER_) / 256, 256, 0, stream>>>(Wq, WqT, DIM_, INNER_);
  conv_transpose_kernel<<<(DIM_VIS_ * 2 * INNER_) / 256, 256, 0, stream>>>(
      Wkv, WkvT, DIM_VIS_, 2 * INNER_);
  conv_transpose_kernel<<<(INNER_ * DIM_) / 256, 256, 0, stream>>>(Wout, WoutT, INNER_, DIM_);
  scan_kernel<<<B_, 1024, 0, stream>>>(mloc, tt);

  // q = LN(x) @ Wq          : M=8192 N=512  K=2048
  gemm_bf16_wmma<<<dim3(INNER_ / 128, ROWS_ / 128), 128, 0, stream>>>(
      xn, WqT, qbuf, ROWS_, INNER_, DIM_);
  // kv = media_f @ Wkv      : M=2048 N=1024 K=1024
  gemm_bf16_wmma<<<dim3((2 * INNER_) / 128, (B_ * JKEYS_) / 128), 128, 0, stream>>>(
      mediab, WkvT, kvbuf, B_ * JKEYS_, 2 * INNER_, DIM_VIS_);
  // masked softmax attention (64 keys per query)
  attn_core_kernel<<<(ROWS_ * HEADS_) / 8, 256, 0, stream>>>(qbuf, kvbuf, tt, ao);
  // out = attn_out @ Wout   : M=8192 N=2048 K=512
  gemm_bf16_wmma<<<dim3(DIM_ / 128, ROWS_ / 128), 128, 0, stream>>>(
      ao, WoutT, out, ROWS_, DIM_, INNER_);
}
